// BodyNet_24386824307416
// MI455X (gfx1250) — matrix operand order repaired and checked by reading, not compiled, not run on hardware
//
#include <hip/hip_runtime.h>
#include <hip/hip_bf16.h>
#include <cstdint>

// ---------------- types ----------------
typedef __attribute__((ext_vector_type(16))) __bf16 v16bf;
typedef __attribute__((ext_vector_type(8)))  __bf16 v8bf;
typedef __attribute__((ext_vector_type(8)))  float  v8f;

#define N_BODIES 16
#define NG       17
#define BATCH    4096
#define M_ROWS   (BATCH * NG)   // 69632
#define D_OBS    11
#define EMB      128

// ---------------- workspace layout (bytes) ----------------
// x32   : 69632*256*4 = 71,303,168
// x16   : 69632*256*2 = 35,651,584
// XL16  : 35,651,584
// XR16  : 35,651,584
// WT    : 6 * 256*256*2 = 786,432
// total ~179 MB
static constexpr size_t OFF_X16 = 71303168;
static constexpr size_t OFF_XL  = OFF_X16 + 35651584;
static constexpr size_t OFF_XR  = OFF_XL  + 35651584;
static constexpr size_t OFF_WT  = OFF_XR  + 35651584;
static constexpr size_t WT_SLOT = 256 * 256 * 2;

// ---------------- static edge graph (from _build_edges) ----------------
// e = 2i   (i=0..14): src=i,   dst=i+1
// e = 2i+1          : src=i+1, dst=i
// e = 30+2i (i=0..15): src=16, dst=i
// e = 31+2i          : src=i,  dst=16
__device__ __forceinline__ int esrc(int e) {
  if (e < 30) { int i = e >> 1; return (e & 1) ? i + 1 : i; }
  int k = e - 30; return (k & 1) ? (k >> 1) : 16;
}
__device__ __forceinline__ int edst(int e) {
  if (e < 30) { int i = e >> 1; return (e & 1) ? i : i + 1; }
  int k = e - 30; return (k & 1) ? 16 : (k >> 1);
}
// incoming edge list per destination node
__device__ __forceinline__ int incoming(int d, int* el) {
  if (d < 16) {
    int n = 0;
    if (d > 0)  el[n++] = 2 * (d - 1);   // d-1 -> d
    if (d < 15) el[n++] = 2 * d + 1;     // d+1 -> d
    el[n++] = 30 + 2 * d;                // 16  -> d
    return n;
  }
  for (int i = 0; i < 16; ++i) el[i] = 31 + 2 * i; // i -> 16
  return 16;
}

// ---------------- weight transpose + bf16 convert ----------------
// in: (K x N) f32 row-major -> out: (N x K) bf16 row-major
__global__ __launch_bounds__(256) void transpose_bf16_kernel(
    const float* __restrict__ in, __bf16* __restrict__ out, int K, int N) {
  int idx = blockIdx.x * 256 + threadIdx.x;
  if (idx >= K * N) return;
  int k = idx / N, n = idx % N;
  out[n * K + k] = (__bf16)in[idx];
}

// ---------------- tokenizer: obs -> x (B,17,128) f32 + bf16 ----------------
__global__ __launch_bounds__(128) void tokenize_kernel(
    const float* __restrict__ obs, const float* __restrict__ tokW,
    const float* __restrict__ tokb, float* __restrict__ x32,
    __bf16* __restrict__ x16) {
  __shared__ float s_obs[N_BODIES * D_OBS];
  int b = blockIdx.x;
  for (int i = threadIdx.x; i < N_BODIES * D_OBS; i += 128)
    s_obs[i] = obs[(size_t)b * (N_BODIES * D_OBS) + i];
  __syncthreads();
  int e = threadIdx.x;  // channel 0..127
  float vsum = 0.f;
  for (int n = 0; n < N_BODIES; ++n) {
    float acc = tokb[n * EMB + e];
    for (int d = 0; d < D_OBS; ++d)
      acc += s_obs[n * D_OBS + d] * tokW[(n * D_OBS + d) * EMB + e];
    vsum += acc;
    size_t idx = ((size_t)b * NG + n) * EMB + e;
    x32[idx] = acc; x16[idx] = (__bf16)acc;
  }
  float v = vsum * (1.f / 16.f);
  size_t idx = ((size_t)b * NG + N_BODIES) * EMB + e;
  x32[idx] = v; x16[idx] = (__bf16)v;
}

// ---------------- WMMA GEMM: Y(M,N) = X(M,K) @ WT(N,K)^T + bias ----------------
// block = 256 threads = 8 waves; wave tile = 32M x 64N; K step 32 (bf16 WMMA).
__global__ __launch_bounds__(256) void gemm_bias_kernel(
    const __bf16* __restrict__ X, const __bf16* __restrict__ WT,
    const float* __restrict__ bias, __bf16* __restrict__ Y, int N, int K) {
  int lane = threadIdx.x & 31;
  int wave = threadIdx.x >> 5;
  int l  = lane & 15;
  int hi = lane >> 4;
  int mr  = (blockIdx.x * 8 + wave) * 32;
  int nc0 = blockIdx.y * 64;

  v8f acc[2][4] = {};  // zero-init

  for (int kb = 0; kb < K; kb += 32) {
    // A tiles (16x32 bf16): lane<16 holds row l, K = kb+{0..15}; lane>=16 K+16
    v16bf a[2];
#pragma unroll
    for (int i = 0; i < 2; ++i) {
      const __bf16* ap = X + (size_t)(mr + i * 16 + l) * K + kb + hi * 16;
      v8bf a0 = *(const v8bf*)(ap);
      v8bf a1 = *(const v8bf*)(ap + 8);
      a[i] = __builtin_shufflevector(a0, a1, 0, 1, 2, 3, 4, 5, 6, 7,
                                             8, 9, 10, 11, 12, 13, 14, 15);
    }
    // B tiles (32x16 bf16): col = nc+l, K = kb + hi*16 + e (contig. in WT row)
#pragma unroll
    for (int t = 0; t < 4; ++t) {
      const __bf16* bp = WT + (size_t)(nc0 + t * 16 + l) * K + kb + hi * 16;
      v16bf bt = *(const v16bf*)bp;
      acc[0][t] = __builtin_amdgcn_wmma_f32_16x16x32_bf16(
          false, a[0], false, bt, (short)0, acc[0][t], false, false);
      acc[1][t] = __builtin_amdgcn_wmma_f32_16x16x32_bf16(
          false, a[1], false, bt, (short)0, acc[1][t], false, false);
    }
  }
  // D layout: VGPR j -> row = mr + hi*8 + j, col = nc + l
#pragma unroll
  for (int t = 0; t < 4; ++t) {
    int col = nc0 + t * 16 + l;
    float bv = bias[col];
#pragma unroll
    for (int i = 0; i < 2; ++i) {
#pragma unroll
      for (int j = 0; j < 8; ++j) {
        int row = mr + i * 16 + hi * 8 + j;
        Y[(size_t)row * N + col] = (__bf16)(acc[i][t][j] + bv);
      }
    }
  }
}

// ---------------- GATv2 edge + softmax + aggregate + LN (+GELU/residual) ----
// One wave per batch sample; 8 samples per 256-thread block.
template <int H, int C, bool APPLY_GELU, bool RES>
__global__ __launch_bounds__(256) void gat_edge_kernel(
    const __bf16* __restrict__ XL, const __bf16* __restrict__ XR,
    const float* __restrict__ We, const float* __restrict__ att,
    const float* __restrict__ bias, const float* __restrict__ lng,
    const float* __restrict__ lnb, float* __restrict__ x32,
    __bf16* __restrict__ x16) {
  constexpr int HC  = H * C;
  constexpr int NCH = HC / 32;
  __shared__ float s_alpha[8][62][4];
  __shared__ float s_m[8][17][4];
  __shared__ float s_s[8][17][4];

  int lane = threadIdx.x & 31;
  int wave = threadIdx.x >> 5;
  int b = blockIdx.x * 8 + wave;
  const __bf16* xlb = XL + (size_t)b * NG * HC;
  const __bf16* xrb = XR + (size_t)b * NG * HC;

  // phase 1: per-edge per-head logits (lanes split channels, butterfly reduce)
  for (int e = 0; e < 62; ++e) {
    int s = esrc(e), d = edst(e);
    const __bf16* xs = xlb + s * HC;
    const __bf16* xd = xrb + d * HC;
    for (int hh = 0; hh < H; ++hh) {
      float p = 0.f;
      for (int j = lane; j < C; j += 32) {
        int ch = hh * C + j;
        float t = (float)xs[ch] + (float)xd[ch] + We[ch];
        t = t > 0.f ? t : 0.2f * t;  // leaky_relu(0.2)
        p += t * att[ch];
      }
      for (int off = 16; off > 0; off >>= 1) p += __shfl_xor(p, off, 32);
      if (lane == 0) s_alpha[wave][e][hh] = p;
    }
  }
  __syncthreads();

  // phase 2: per-(dst,head) max and exp-sum
  for (int p = lane; p < 17 * H; p += 32) {
    int d = p / H, hh = p % H;
    int el[16]; int ne = incoming(d, el);
    float m = -1e30f;
    for (int k = 0; k < ne; ++k) m = fmaxf(m, s_alpha[wave][el[k]][hh]);
    float s = 0.f;
    for (int k = 0; k < ne; ++k) s += __expf(s_alpha[wave][el[k]][hh] - m);
    s_m[wave][d][hh] = m; s_s[wave][d][hh] = s;
  }
  __syncthreads();
  // logits -> alpha in place
  for (int q = lane; q < 62 * H; q += 32) {
    int e = q / H, hh = q % H;
    int d = edst(e);
    s_alpha[wave][e][hh] =
        __expf(s_alpha[wave][e][hh] - s_m[wave][d][hh]) / s_s[wave][d][hh];
  }
  __syncthreads();

  // phase 3: aggregate, +bias, LayerNorm, (GELU), (residual), store f32+bf16
  for (int d = 0; d < NG; ++d) {
    float acc[NCH];
#pragma unroll
    for (int u = 0; u < NCH; ++u) acc[u] = bias[lane + 32 * u];
    int el[16]; int ne = incoming(d, el);
    for (int k = 0; k < ne; ++k) {
      int e = el[k]; int s = esrc(e);
      const __bf16* xs = xlb + s * HC;
#pragma unroll
      for (int u = 0; u < NCH; ++u) {
        int ch = lane + 32 * u;
        acc[u] += s_alpha[wave][e][ch / C] * (float)xs[ch];
      }
    }
    float sum = 0.f, sq = 0.f;
#pragma unroll
    for (int u = 0; u < NCH; ++u) { sum += acc[u]; sq += acc[u] * acc[u]; }
    for (int off = 16; off > 0; off >>= 1) {
      sum += __shfl_xor(sum, off, 32);
      sq  += __shfl_xor(sq,  off, 32);
    }
    float mu  = sum * (1.f / HC);
    float var = sq * (1.f / HC) - mu * mu;
    float inv = rsqrtf(var + 1e-5f);
    size_t base = ((size_t)b * NG + d) * HC;
#pragma unroll
    for (int u = 0; u < NCH; ++u) {
      int ch = lane + 32 * u;
      float y = (acc[u] - mu) * inv * lng[ch] + lnb[ch];
      if (APPLY_GELU) y = 0.5f * y * (1.f + erff(y * 0.70710678118654752f));
      if (RES) y += x32[base + ch];
      x32[base + ch] = y;
      x16[base + ch] = (__bf16)y;
    }
  }
}

// ---------------- detokenizer: act = x[:, :16] @ detok_W + detok_b ----------
__global__ __launch_bounds__(64) void detok_kernel(
    const float* __restrict__ x32, const float* __restrict__ dW,
    const float* __restrict__ db, float* __restrict__ out) {
  int b = blockIdx.x;
  int t = threadIdx.x;
  if (t >= N_BODIES * 3) return;
  int n = t / 3, dd = t % 3;
  const float* xr = x32 + ((size_t)b * NG + n) * EMB;
  float acc = db[n * 3 + dd];
  for (int e = 0; e < EMB; ++e) acc += xr[e] * dW[(n * EMB + e) * 3 + dd];
  out[(size_t)b * (N_BODIES * 3) + t] = acc;
}

// ---------------- launch ----------------
extern "C" void kernel_launch(void* const* d_in, const int* in_sizes, int n_in,
                              void* d_out, int out_size, void* d_ws,
                              size_t ws_size, hipStream_t stream) {
  (void)in_sizes; (void)n_in; (void)out_size; (void)ws_size;

  // input order (setup_inputs insertion order, nested structures flattened):
  const float* obs  = (const float*)d_in[0];
  const float* tokW = (const float*)d_in[1];
  const float* tokb = (const float*)d_in[2];
  const int FIN[3] = {128, 256, 256};
  const int HCv[3] = {256, 256, 128};
  const float *Wl[3], *bl[3], *Wr[3], *br[3], *We[3], *att[3], *bias[3];
  for (int L = 0; L < 3; ++L) {
    int base = 3 + 7 * L;
    Wl[L]   = (const float*)d_in[base + 0];
    bl[L]   = (const float*)d_in[base + 1];
    Wr[L]   = (const float*)d_in[base + 2];
    br[L]   = (const float*)d_in[base + 3];
    We[L]   = (const float*)d_in[base + 4];
    att[L]  = (const float*)d_in[base + 5];
    bias[L] = (const float*)d_in[base + 6];
  }
  const float *lng[3], *lnb[3];
  for (int L = 0; L < 3; ++L) {
    lng[L] = (const float*)d_in[24 + 2 * L];
    lnb[L] = (const float*)d_in[25 + 2 * L];
  }
  const float* dW = (const float*)d_in[30];
  const float* db = (const float*)d_in[31];
  // d_in[32], d_in[33] = edge_src/dst (static graph, hardcoded in kernels)
  float* out = (float*)d_out;

  char*   ws  = (char*)d_ws;
  float*  x32 = (float*)(ws);
  __bf16* x16 = (__bf16*)(ws + OFF_X16);
  __bf16* XL  = (__bf16*)(ws + OFF_XL);
  __bf16* XR  = (__bf16*)(ws + OFF_XR);
  __bf16* WlT[3]; __bf16* WrT[3];
  for (int L = 0; L < 3; ++L) {
    WlT[L] = (__bf16*)(ws + OFF_WT + (size_t)(2 * L) * WT_SLOT);
    WrT[L] = (__bf16*)(ws + OFF_WT + (size_t)(2 * L + 1) * WT_SLOT);
  }

  // weight prep: f32 (K,N) -> bf16 (N,K)
  for (int L = 0; L < 3; ++L) {
    int kn = FIN[L] * HCv[L];
    int g  = (kn + 255) / 256;
    transpose_bf16_kernel<<<g, 256, 0, stream>>>(Wl[L], WlT[L], FIN[L], HCv[L]);
    transpose_bf16_kernel<<<g, 256, 0, stream>>>(Wr[L], WrT[L], FIN[L], HCv[L]);
  }

  tokenize_kernel<<<BATCH, 128, 0, stream>>>(obs, tokW, tokb, x32, x16);

  const int GEMM_GX = M_ROWS / 256;  // 272

  // ---- layer 0: fin=128, hc=256, h=4, c=64, gelu, no residual ----
  gemm_bias_kernel<<<dim3(GEMM_GX, HCv[0] / 64), 256, 0, stream>>>(
      x16, WlT[0], bl[0], XL, HCv[0], FIN[0]);
  gemm_bias_kernel<<<dim3(GEMM_GX, HCv[0] / 64), 256, 0, stream>>>(
      x16, WrT[0], br[0], XR, HCv[0], FIN[0]);
  gat_edge_kernel<4, 64, true, false><<<BATCH / 8, 256, 0, stream>>>(
      XL, XR, We[0], att[0], bias[0], lng[0], lnb[0], x32, x16);

  // ---- layer 1: fin=256, hc=256, h=4, c=64, gelu, residual ----
  gemm_bias_kernel<<<dim3(GEMM_GX, HCv[1] / 64), 256, 0, stream>>>(
      x16, WlT[1], bl[1], XL, HCv[1], FIN[1]);
  gemm_bias_kernel<<<dim3(GEMM_GX, HCv[1] / 64), 256, 0, stream>>>(
      x16, WrT[1], br[1], XR, HCv[1], FIN[1]);
  gat_edge_kernel<4, 64, true, true><<<BATCH / 8, 256, 0, stream>>>(
      XL, XR, We[1], att[1], bias[1], lng[1], lnb[1], x32, x16);

  // ---- layer 2: fin=256, hc=128, h=1, c=128, no gelu, no residual ----
  gemm_bias_kernel<<<dim3(GEMM_GX, HCv[2] / 64), 256, 0, stream>>>(
      x16, WlT[2], bl[2], XL, HCv[2], FIN[2]);
  gemm_bias_kernel<<<dim3(GEMM_GX, HCv[2] / 64), 256, 0, stream>>>(
      x16, WrT[2], br[2], XR, HCv[2], FIN[2]);
  gat_edge_kernel<1, 128, false, false><<<BATCH / 8, 256, 0, stream>>>(
      XL, XR, We[2], att[2], bias[2], lng[2], lnb[2], x32, x16);

  detok_kernel<<<BATCH, 64, 0, stream>>>(x32, dW, db, out);
}
